// LlamaAttention_heavy_hitter_42477226557700
// MI455X (gfx1250) — compile-verified, matrix operand order checked by reading, not run on hardware
//
#include <hip/hip_runtime.h>

typedef __attribute__((ext_vector_type(16))) _Float16 v16h;
typedef __attribute__((ext_vector_type(8)))  _Float16 v8h;
typedef __attribute__((ext_vector_type(8)))  float    v8f;
typedef __attribute__((ext_vector_type(4)))  int      v4i;

#define B_    2
#define S_    1024
#define H_    16
#define D_    128
#define HID_  2048
#define Z_    (B_*H_)      // 32 batched heads
#define CACHEB 204
#define RECENT 102
#define NKEEP  6
#define LDSP  40           // padded LDS row pitch in halves (80B = 20 banks)

enum { OUT_F32_RM = 0, OUT_F16_BHSD = 1, OUT_F16_BSHD = 2 };

#if __has_builtin(__builtin_amdgcn_global_load_async_to_lds_b128)
#define HAVE_ASYNC_LDS 1
#else
#define HAVE_ASYNC_LDS 0
#endif

__device__ __forceinline__ void stage16(const _Float16* g, _Float16* l)
{
#if HAVE_ASYNC_LDS
  __builtin_amdgcn_global_load_async_to_lds_b128(
      (__attribute__((address_space(1))) v4i*)(void*)g,
      (__attribute__((address_space(3))) v4i*)(void*)l, 0, 0);
#else
  *(v8h*)l = *(const v8h*)g;
#endif
}

__device__ __forceinline__ void wait_stage()
{
#if HAVE_ASYNC_LDS
#if __has_builtin(__builtin_amdgcn_s_wait_asynccnt)
  __builtin_amdgcn_s_wait_asynccnt(0);
#else
  asm volatile("s_wait_asynccnt 0x0" ::: "memory");
#endif
#endif
}

// ---------------------------------------------------------------------------
// WMMA GEMM with async-LDS tile staging:
//   C[row,col] = scale * sum_k A[row,k] * Bt[col,k]
// A: [M,K] row-major f16, Bt: [N,K] row-major f16 (B pre-transposed).
// 256 threads = 8 waves; block tile 128x128; wave tile 32x64 = 2x4 WMMA tiles.
// Per k-chunk (32): stage A-tile/B-tile (128x32 halves each) to LDS via
// global_load_async_to_lds_b128, wait on ASYNCcnt, then ds_load_b128
// fragment reads in the ISA 7.12.2 register layout.
// ---------------------------------------------------------------------------
template <int OUTMODE>
__global__ __launch_bounds__(256) void wmma_gemm_kernel(
    const _Float16* __restrict__ A, long aBatch, int lda,
    const _Float16* __restrict__ Bt, long bBatch, int ldb,
    void* __restrict__ Cv, long cBatch, int ldc,
    int K, float scale)
{
  const int z = blockIdx.z;
  A  += (long)z * aBatch;
  Bt += (long)z * bBatch;

  const int tid    = threadIdx.x;
  const int wid    = tid >> 5;
  const int lane   = tid & 31;
  const int laneLo = lane & 15;   // m (A) / n (B) / col (C) within 16x16 tile
  const int laneHi = lane >> 4;   // selects K-subgroup for A/B, +8 rows for C

  const int rowBlk = blockIdx.x * 128;
  const int colBlk = blockIdx.y * 128;
  const int rowBase = (wid & 3) * 32;   // 4 waves along M (block-local)
  const int colBase = (wid >> 2) * 64;  // 2 waves along N (block-local)

  __shared__ _Float16 lA[128 * LDSP];
  __shared__ _Float16 lB[128 * LDSP];

  v8f acc[2][4];
#pragma unroll
  for (int i = 0; i < 2; ++i)
#pragma unroll
    for (int j = 0; j < 4; ++j)
#pragma unroll
      for (int r = 0; r < 8; ++r) acc[i][j][r] = 0.0f;

  for (int k0 = 0; k0 < K; k0 += 32) {
    // --- stage 128x32 A and B tiles: 512 16B chunks each, 2+2 per thread ---
#pragma unroll
    for (int c = tid; c < 512; c += 256) {
      const int r = c >> 2, o = (c & 3) * 8;
      const _Float16* ga = A  + (long)(rowBlk + r) * lda + k0 + o;
      const _Float16* gb = Bt + (long)(colBlk + r) * ldb + k0 + o;
      stage16(ga, &lA[r * LDSP + o]);
      stage16(gb, &lB[r * LDSP + o]);
      __builtin_prefetch(ga + 32, 0, 3);   // near-scope prefetch, next k-chunk
      __builtin_prefetch(gb + 32, 0, 3);
    }
    wait_stage();
    __syncthreads();

    // --- fragment reads (ds_load_b128) + 8 WMMAs ---
    v16h af[2], bf[4];
#pragma unroll
    for (int i = 0; i < 2; ++i) {
      const _Float16* p = &lA[(rowBase + i * 16 + laneLo) * LDSP + laneHi * 8];
      v8h lo = *(const v8h*)(p);
      v8h hi = *(const v8h*)(p + 16);
#pragma unroll
      for (int e = 0; e < 8; ++e) { af[i][e] = lo[e]; af[i][e + 8] = hi[e]; }
    }
#pragma unroll
    for (int j = 0; j < 4; ++j) {
      const _Float16* p = &lB[(colBase + j * 16 + laneLo) * LDSP + laneHi * 8];
      v8h lo = *(const v8h*)(p);
      v8h hi = *(const v8h*)(p + 16);
#pragma unroll
      for (int e = 0; e < 8; ++e) { bf[j][e] = lo[e]; bf[j][e + 8] = hi[e]; }
    }
#pragma unroll
    for (int i = 0; i < 2; ++i)
#pragma unroll
      for (int j = 0; j < 4; ++j)
        acc[i][j] = __builtin_amdgcn_wmma_f32_16x16x32_f16(
            false, af[i], false, bf[j], (short)0, acc[i][j], false, false);
    __syncthreads();
  }

#pragma unroll
  for (int i = 0; i < 2; ++i)
#pragma unroll
    for (int j = 0; j < 4; ++j) {
      const int col = colBlk + colBase + j * 16 + laneLo;
#pragma unroll
      for (int r = 0; r < 8; ++r) {
        const int row = rowBlk + rowBase + i * 16 + laneHi * 8 + r;
        const float v = acc[i][j][r] * scale;
        if (OUTMODE == OUT_F32_RM) {
          ((float*)Cv)[(long)z * cBatch + (long)row * ldc + col] = v;
        } else if (OUTMODE == OUT_F16_BHSD) {
          // row = b*S + s  (M = B*S), col = h*D + d  -> [B,H,S,D]
          const int b = row >> 10, s = row & (S_ - 1);
          const int h = col >> 7,  d = col & (D_ - 1);
          ((_Float16*)Cv)[(((long)(b * H_ + h) * S_) + s) * D_ + d] = (_Float16)v;
        } else { // OUT_F16_BSHD: z = b*H + h, row = t, col = d -> [B,S,H*D]
          const int b = z >> 4, h = z & (H_ - 1);
          ((_Float16*)Cv)[((long)(b * S_ + row) * HID_) + h * D_ + col] = (_Float16)v;
        }
      }
    }
}

// ---------------------------------------------------------------------------
// Elementwise helpers
// ---------------------------------------------------------------------------
__global__ void f32_to_f16_kernel(const float* __restrict__ in,
                                  _Float16* __restrict__ out, long n)
{
  long i = (long)blockIdx.x * blockDim.x + threadIdx.x;
  if (i < n) out[i] = (_Float16)in[i];
}

// W [K,N] f32 -> Wt [N,K] f16 so GEMM B-fragments are contiguous 16B loads.
__global__ void transpose_w_kernel(const float* __restrict__ W,
                                   _Float16* __restrict__ Wt)
{
  long i = (long)blockIdx.x * blockDim.x + threadIdx.x;
  if (i >= (long)HID_ * HID_) return;
  int k = (int)(i >> 11), n = (int)(i & (HID_ - 1));
  Wt[(long)n * HID_ + k] = (_Float16)W[i];
}

// V [B,H,S,D] -> Vt [B,H,D,S] (f16) for the PV GEMM's B operand.
__global__ void transpose_v_kernel(const _Float16* __restrict__ V,
                                   _Float16* __restrict__ Vt)
{
  long i = (long)blockIdx.x * blockDim.x + threadIdx.x;
  if (i >= (long)Z_ * S_ * D_) return;
  long z = i >> 17;                 // S_*D_ = 131072
  int  r = (int)(i & (S_ * D_ - 1));
  int  s = r >> 7, d = r & (D_ - 1);
  Vt[z * (S_ * D_) + (long)d * S_ + s] = V[i];
}

// RoPE in-place on Q,K (f16 BHSD). One thread per (z, s, freq-pair).
__global__ void rope_kernel(_Float16* __restrict__ Q, _Float16* __restrict__ K)
{
  long gid = (long)blockIdx.x * blockDim.x + threadIdx.x;
  if (gid >= (long)Z_ * S_ * (D_ / 2)) return;
  int  i  = (int)(gid & 63);         // freq index 0..63
  long zs = gid >> 6;                // z*S + s
  int  s  = (int)(zs & (S_ - 1));
  float f   = __expf(-(float)i * (9.210340371976184f / 64.0f)); // 10000^(-i/64)
  float ang = (float)s * f;
  float c = __cosf(ang), sn = __sinf(ang);
  _Float16* q = Q + zs * D_;
  _Float16* k = K + zs * D_;
  float q0 = (float)q[i], q1 = (float)q[i + 64];
  q[i]      = (_Float16)(q0 * c - q1 * sn);
  q[i + 64] = (_Float16)(q1 * c + q0 * sn);
  float k0 = (float)k[i], k1 = (float)k[i + 64];
  k[i]      = (_Float16)(k0 * c - k1 * sn);
  k[i + 64] = (_Float16)(k1 * c + k0 * sn);
}

// Top-6 |k|-mean dims per (b,h) from first CACHEB (roped) keys.
// Tie-break: larger index kept (matches double-argsort rank semantics).
__global__ __launch_bounds__(128) void kmask_kernel(
    const _Float16* __restrict__ K, int* __restrict__ keptIdx)
{
  const int z = blockIdx.x, d = threadIdx.x;
  __shared__ float ma[D_];
  const _Float16* kp = K + (long)z * S_ * D_;
  float acc = 0.f;
  for (int t = 0; t < CACHEB; ++t) acc += fabsf((float)kp[t * D_ + d]);
  ma[d] = acc * (1.0f / CACHEB);
  __syncthreads();
  if (d == 0) {
    for (int i = 0; i < NKEEP; ++i) {
      float best = -1.f; int bi = 0;
      for (int j = 0; j < D_; ++j)
        if (ma[j] >= best) { best = ma[j]; bi = j; }  // >= : larger idx wins tie
      keptIdx[z * NKEEP + i] = bi;
      ma[bi] = -2.f;
    }
  }
}

// ---------------------------------------------------------------------------
// Serial H2O scan per (b,h). Eviction argmin only ever reads FULL scores of
// non-evicted keys (evicted ones carry hist=inf in the reference), so the
// whole scan runs off the precomputed f32 score matrix (L2-resident).
// Records evict_step[s] (INT_MAX if never evicted).
// ---------------------------------------------------------------------------
__global__ __launch_bounds__(256) void h2o_scan_kernel(
    const float* __restrict__ scores, int* __restrict__ evictStep)
{
  const int z = blockIdx.x;
  const float* Sz = scores + (long)z * S_ * S_;
  int* ev = evictStep + z * S_;
  __shared__ unsigned evbits[S_ / 32];
  __shared__ float rmin[256];
  __shared__ int   ridx[256];
  const int tid = threadIdx.x;
  for (int s = tid; s < S_; s += 256) ev[s] = 0x7fffffff;
  if (tid < S_ / 32) evbits[tid] = 0u;
  __syncthreads();

  for (int t = CACHEB + 1; t < S_; ++t) {
    const int limit = t - RECENT;
    const float* row = Sz + (long)(t - 1) * S_;
    float best = 3.4e38f; int bidx = 0x7fffffff;
    for (int s = tid; s < limit; s += 256) {
      if (evbits[s >> 5] & (1u << (s & 31))) continue;
      float v = row[s];
      if (v < best || (v == best && s < bidx)) { best = v; bidx = s; }
    }
    rmin[tid] = best; ridx[tid] = bidx;
    __syncthreads();
    for (int off = 128; off > 0; off >>= 1) {
      if (tid < off) {
        float v = rmin[tid + off]; int ix = ridx[tid + off];
        if (v < rmin[tid] || (v == rmin[tid] && ix < ridx[tid])) {
          rmin[tid] = v; ridx[tid] = ix;
        }
      }
      __syncthreads();
    }
    if (tid == 0) {
      int s = ridx[0];
      evbits[s >> 5] |= (1u << (s & 31));
      ev[s] = t;
    }
    __syncthreads();
  }
}

// ---------------------------------------------------------------------------
// Row softmax. Evicted keys (evict_step <= t) use the 6-kept-dim score,
// recomputed on the fly; everything else uses the WMMA full score.
// ---------------------------------------------------------------------------
__global__ __launch_bounds__(256) void softmax_probs_kernel(
    const float* __restrict__ scores, const _Float16* __restrict__ Q,
    const _Float16* __restrict__ K, const int* __restrict__ keptIdx,
    const int* __restrict__ evictStep, _Float16* __restrict__ probs)
{
  const int t = blockIdx.x, z = blockIdx.y, tid = threadIdx.x;
  __shared__ float rowv[S_];
  __shared__ float red[256];
  __shared__ int   kd[NKEEP];
  __shared__ float qv[NKEEP];
  const float scale = 0.08838834764831845f;   // 1/sqrt(128)
  if (tid < NKEEP) {
    kd[tid] = keptIdx[z * NKEEP + tid];
    qv[tid] = (float)Q[(long)z * S_ * D_ + (long)t * D_ + kd[tid]];
  }
  __syncthreads();

  const float* row = scores + ((long)z * S_ + t) * S_;
  const int* ev = evictStep + z * S_;
  const _Float16* Kz = K + (long)z * S_ * D_;

  float lmax = -3.4e38f;
  for (int s = tid; s <= t; s += 256) {
    float v;
    if (ev[s] <= t) {
      const _Float16* kp = Kz + (long)s * D_;
      float a = 0.f;
#pragma unroll
      for (int i = 0; i < NKEEP; ++i) a += qv[i] * (float)kp[kd[i]];
      v = a * scale;
    } else {
      v = row[s];
    }
    rowv[s] = v;
    lmax = fmaxf(lmax, v);
  }
  red[tid] = lmax;
  __syncthreads();
  for (int off = 128; off > 0; off >>= 1) {
    if (tid < off) red[tid] = fmaxf(red[tid], red[tid + off]);
    __syncthreads();
  }
  const float m = red[0];
  __syncthreads();
  float lsum = 0.f;
  for (int s = tid; s <= t; s += 256) {
    float e = __expf(rowv[s] - m);
    rowv[s] = e; lsum += e;
  }
  red[tid] = lsum;
  __syncthreads();
  for (int off = 128; off > 0; off >>= 1) {
    if (tid < off) red[tid] += red[tid + off];
    __syncthreads();
  }
  const float inv = 1.0f / red[0];
  _Float16* prow = probs + ((long)z * S_ + t) * S_;
  for (int s = tid; s < S_; s += 256)
    prow[s] = (s <= t) ? (_Float16)(rowv[s] * inv) : (_Float16)0.0f;
}

// ---------------------------------------------------------------------------
extern "C" void kernel_launch(void* const* d_in, const int* in_sizes, int n_in,
                              void* d_out, int out_size, void* d_ws, size_t ws_size,
                              hipStream_t stream)
{
  (void)in_sizes; (void)n_in; (void)out_size; (void)ws_size;
  const float* hs = (const float*)d_in[0];
  const float* wq = (const float*)d_in[1];
  const float* wk = (const float*)d_in[2];
  const float* wv = (const float*)d_in[3];
  const float* wo = (const float*)d_in[4];
  // d_in[5] = attention_mask: pure causal; applied analytically in softmax.

  char* w = (char*)d_ws;
  auto alloc = [&](size_t bytes) { char* p = w; w += (bytes + 255) & ~(size_t)255; return p; };
  const size_t MAT = (size_t)HID_ * HID_;        // 4M elems
  const size_t BHSD = (size_t)Z_ * S_ * D_;      // 4M elems
  _Float16* X16  = (_Float16*)alloc(MAT * 2);
  _Float16* WqT  = (_Float16*)alloc(MAT * 2);
  _Float16* WkT  = (_Float16*)alloc(MAT * 2);
  _Float16* WvT  = (_Float16*)alloc(MAT * 2);
  _Float16* WoT  = (_Float16*)alloc(MAT * 2);
  _Float16* Q16  = (_Float16*)alloc(BHSD * 2);
  _Float16* K16  = (_Float16*)alloc(BHSD * 2);
  _Float16* V16  = (_Float16*)alloc(BHSD * 2);
  _Float16* Vt16 = (_Float16*)alloc(BHSD * 2);
  float*    SC   = (float*)   alloc((size_t)Z_ * S_ * S_ * 4);   // 128 MB
  _Float16* P16  = (_Float16*)alloc((size_t)Z_ * S_ * S_ * 2);   // 64 MB
  _Float16* AO16 = (_Float16*)alloc(MAT * 2);
  int*      KIDX = (int*)     alloc((size_t)Z_ * NKEEP * 4);
  int*      EVST = (int*)     alloc((size_t)Z_ * S_ * 4);

  const dim3 blk256(256);

  // 1) f32 -> f16 packing / weight transposes
  f32_to_f16_kernel<<<(MAT + 255) / 256, blk256, 0, stream>>>(hs, X16, (long)MAT);
  transpose_w_kernel<<<(MAT + 255) / 256, blk256, 0, stream>>>(wq, WqT);
  transpose_w_kernel<<<(MAT + 255) / 256, blk256, 0, stream>>>(wk, WkT);
  transpose_w_kernel<<<(MAT + 255) / 256, blk256, 0, stream>>>(wv, WvT);
  transpose_w_kernel<<<(MAT + 255) / 256, blk256, 0, stream>>>(wo, WoT);

  // 2) Q/K/V projections (WMMA), scattered into [B,H,S,D] f16
  const dim3 gProj(16, 16, 1);
  wmma_gemm_kernel<OUT_F16_BHSD><<<gProj, blk256, 0, stream>>>(
      X16, 0, HID_, WqT, 0, HID_, Q16, 0, 0, HID_, 1.0f);
  wmma_gemm_kernel<OUT_F16_BHSD><<<gProj, blk256, 0, stream>>>(
      X16, 0, HID_, WkT, 0, HID_, K16, 0, 0, HID_, 1.0f);
  wmma_gemm_kernel<OUT_F16_BHSD><<<gProj, blk256, 0, stream>>>(
      X16, 0, HID_, WvT, 0, HID_, V16, 0, 0, HID_, 1.0f);

  // 3) RoPE on Q,K; kept-dim selection; V transpose
  rope_kernel<<<((long)Z_ * S_ * 64 + 255) / 256, blk256, 0, stream>>>(Q16, K16);
  kmask_kernel<<<Z_, dim3(128), 0, stream>>>(K16, KIDX);
  transpose_v_kernel<<<(BHSD + 255) / 256, blk256, 0, stream>>>(V16, Vt16);

  // 4) Full scores = scale * Q K^T per head (WMMA, f32 out; L2-resident)
  const dim3 gScore(8, 8, Z_);
  wmma_gemm_kernel<OUT_F32_RM><<<gScore, blk256, 0, stream>>>(
      Q16, (long)S_ * D_, D_, K16, (long)S_ * D_, D_,
      SC, (long)S_ * S_, S_, D_, 0.08838834764831845f);

  // 5) Serial H2O eviction scan (one WG per head)
  h2o_scan_kernel<<<Z_, blk256, 0, stream>>>(SC, EVST);

  // 6) Softmax with on-the-fly 6-dim evicted scores -> probs f16
  softmax_probs_kernel<<<dim3(S_, Z_), blk256, 0, stream>>>(
      SC, Q16, K16, KIDX, EVST, P16);

  // 7) attn @ V per head (WMMA), scattered into [B,S,HID] f16
  const dim3 gAV(8, 1, Z_);
  wmma_gemm_kernel<OUT_F16_BSHD><<<gAV, blk256, 0, stream>>>(
      P16, (long)S_ * S_, S_, Vt16, (long)S_ * D_, S_,
      AO16, 0, 0, S_, 1.0f);

  // 8) Output projection -> f32 d_out
  wmma_gemm_kernel<OUT_F32_RM><<<gProj, blk256, 0, stream>>>(
      AO16, 0, HID_, WoT, 0, HID_,
      d_out, 0, HID_, HID_, 1.0f);
}
